// EfficientCrossAttention_8839042695250
// MI455X (gfx1250) — matrix operand order repaired and checked by reading, not compiled, not run on hardware
//
#include <hip/hip_runtime.h>
#include <stdint.h>

// ---------------------------------------------------------------------------
// CDNA5 (gfx1250) wave32 WMMA cross-attention.
//   D=1024, H=16, HD=64, B=4, SQ=SK=2048  (hardcoded per reference)
// All matmuls: v_wmma_f32_16x16x32_bf16 (bf16 inputs, f32 accumulate).
// Data movement: GLOBAL_LOAD_ASYNC_TO_LDS_B128 (ASYNCcnt) where tiles are
// pure bf16 copies; register-staged f32->bf16 conversion (double-buffered)
// where conversion is required.
// ---------------------------------------------------------------------------

#ifndef __has_builtin
#define __has_builtin(x) 0
#endif

#if __has_builtin(__builtin_amdgcn_global_load_async_to_lds_b128)
#define HAVE_ASYNC_LDS 1
#else
#define HAVE_ASYNC_LDS 0
#endif

typedef __bf16 v16bf __attribute__((ext_vector_type(16)));
typedef float  v8f   __attribute__((ext_vector_type(8)));
typedef int    async_v4i __attribute__((vector_size(16)));   // builtin's pointee type

union FragBF { v16bf v; uint32_t u[8]; };

__device__ __forceinline__ uint16_t f2bf(float x) {
  union { float f; uint32_t u; } c; c.f = x;
  uint32_t u = c.u;
  u += 0x7FFFu + ((u >> 16) & 1u);          // round-to-nearest-even
  return (uint16_t)(u >> 16);
}

// 16B global->LDS copy: async (ASYNCcnt-tracked DMA into LDS) when available.
__device__ __forceinline__ void lds_copy16(const void* g, void* l) {
#if HAVE_ASYNC_LDS
  __builtin_amdgcn_global_load_async_to_lds_b128((async_v4i*)g, (async_v4i*)l, 0, 0);
#else
  *reinterpret_cast<uint4*>(l) = *reinterpret_cast<const uint4*>(g);
#endif
}

__device__ __forceinline__ void async_wait0() {
#if HAVE_ASYNC_LDS
#if __has_builtin(__builtin_amdgcn_s_wait_asynccnt)
  __builtin_amdgcn_s_wait_asynccnt(0);
#else
  asm volatile("s_wait_asynccnt 0x0" ::: "memory");
#endif
#endif
}

// A-matrix fragment, 16x32 bf16 (ISA 7.12.2 "16-bit A-Matrix 16x32"):
//   lane l: M = rowBase + (l&15); K-half base = (l>=16)?8:0
//   VGPR v: K = kbase + base + (v>=4?16:0) + 2*(v&3), pair {K,K+1}
__device__ __forceinline__ v16bf load_a_frag(const unsigned short* lds, int rowBase,
                                             int pitch, int kbase, int lane) {
  const int m  = rowBase + (lane & 15);
  const int kb = kbase + ((lane >> 4) << 3);
  const unsigned short* base = lds + m * pitch;
  FragBF f;
#pragma unroll
  for (int v = 0; v < 8; ++v) {
    const int k = kb + ((v & 4) << 2) + ((v & 3) << 1);
    f.u[v] = *reinterpret_cast<const uint32_t*>(base + k);
  }
  return f.v;
}

// B-matrix fragment, 32x16 bf16, stored in LDS transposed as [N][K]:
//   lane l: N = colBase + (l&15); K-half base = (l>=16)?8:0
//   VGPR v: K = kbase + base + (v>=4?16:0) + 2*(v&3), pair {K,K+1}
__device__ __forceinline__ v16bf load_b_frag(const unsigned short* lds, int colBase,
                                             int pitch, int kbase, int lane) {
  const int n  = colBase + (lane & 15);
  const int kb = kbase + ((lane >> 4) << 3);
  const unsigned short* base = lds + n * pitch + kb;
  FragBF f;
#pragma unroll
  for (int v = 0; v < 8; ++v)
    f.u[v] = *reinterpret_cast<const uint32_t*>(base + ((v & 4) << 2) + ((v & 3) << 1));
  return f.v;
}

__device__ __forceinline__ v8f wmma_bf16(v16bf a, v16bf b, v8f c) {
  return __builtin_amdgcn_wmma_f32_16x16x32_bf16(false, a, false, b, (short)0, c,
                                                 false, false);
}

// ---------------------------------------------------------------------------
// Projection GEMM: C[M,N] = (A[M,K] * W[K,ldw](cols col_off..) + bias) * scale
//  OUT_MODE 0: bf16 row-major   1: bf16 head-reordered [B,H,S,HD]   2: f32
//  Block tile 128x128x32, 8 waves, wave = 32x64 of C. Double-buffered LDS:
//  bf16 A tiles go async global->LDS; f32 tiles convert through registers,
//  with global loads issued before the barrier to overlap previous compute.
// ---------------------------------------------------------------------------
template <bool A_BF16, int OUT_MODE>
__global__ __launch_bounds__(256) void gemm_proj(
    const void* __restrict__ Aptr, int lda, const float* __restrict__ W, int ldw,
    int col_off, const float* __restrict__ bias, float out_scale,
    void* __restrict__ Cptr, int M, int N, int K, int S, int Hn, int HDn) {
  constexpr int AP = 40;   // As pitch: 80B rows -> 16B-aligned stores
  constexpr int BP = 34;   // Bs pitch: [N][K] transposed
  __shared__ __align__(16) unsigned short As[2][128 * AP];
  __shared__ __align__(16) unsigned short Bs[2][128 * BP];

  const int tid  = threadIdx.x;
  const int lane = tid & 31;
  const int wave = tid >> 5;
  const int wr = wave >> 1;      // 0..3  (32-row band)
  const int wc = wave & 1;       // 0..1  (64-col band)
  const int n0 = blockIdx.x * 128;
  const int m0 = blockIdx.y * 128;
  const int nk = K >> 5;

  v8f acc[2][4];
#pragma unroll
  for (int mi = 0; mi < 2; ++mi)
#pragma unroll
    for (int ni = 0; ni < 4; ++ni) acc[mi][ni] = (v8f)0.0f;

  float4 aR[4];   // f32-A staging registers
  float4 wR[4];   // W staging registers

  auto issueA_async = [&](int kk, unsigned short* buf) {   // bf16 A: DMA to LDS
    if constexpr (A_BF16) {
      const unsigned short* A = (const unsigned short*)Aptr;
#pragma unroll
      for (int i = 0; i < 2; ++i) {
        const int fl = tid + i * 256, row = fl >> 2, c = fl & 3;
        lds_copy16(A + (size_t)(m0 + row) * lda + kk + c * 8, buf + row * AP + c * 8);
      }
    }
  };
  auto loadA_regs = [&](int kk) {                          // f32 A: global->regs
    if constexpr (!A_BF16) {
      const float* A = (const float*)Aptr;
#pragma unroll
      for (int i = 0; i < 4; ++i) {
        const int fl = tid + i * 256, row = fl >> 3, c4 = fl & 7;
        aR[i] = *reinterpret_cast<const float4*>(A + (size_t)(m0 + row) * lda + kk + c4 * 4);
      }
    }
  };
  auto storeA_regs = [&](unsigned short* buf) {            // convert + LDS store
    if constexpr (!A_BF16) {
#pragma unroll
      for (int i = 0; i < 4; ++i) {
        const int fl = tid + i * 256, row = fl >> 3, c4 = fl & 7;
        uint2 p;
        p.x = (uint32_t)f2bf(aR[i].x) | ((uint32_t)f2bf(aR[i].y) << 16);
        p.y = (uint32_t)f2bf(aR[i].z) | ((uint32_t)f2bf(aR[i].w) << 16);
        *reinterpret_cast<uint2*>(buf + row * AP + c4 * 4) = p;
      }
    }
  };
  auto loadW_regs = [&](int kk) {
#pragma unroll
    for (int i = 0; i < 4; ++i) {
      const int fl = tid + i * 256, row = fl >> 5, c4 = fl & 31;
      wR[i] = *reinterpret_cast<const float4*>(
          W + (size_t)(kk + row) * ldw + col_off + n0 + c4 * 4);
    }
  };
  auto storeW_regs = [&](unsigned short* buf) {            // transpose to [N][K]
#pragma unroll
    for (int i = 0; i < 4; ++i) {
      const int fl = tid + i * 256, row = fl >> 5, n = (fl & 31) * 4;
      buf[(n + 0) * BP + row] = f2bf(wR[i].x);
      buf[(n + 1) * BP + row] = f2bf(wR[i].y);
      buf[(n + 2) * BP + row] = f2bf(wR[i].z);
      buf[(n + 3) * BP + row] = f2bf(wR[i].w);
    }
  };

  // prologue: stage tile 0 into buffer 0
  loadA_regs(0);
  issueA_async(0, As[0]);
  loadW_regs(0);
  storeA_regs(As[0]);
  storeW_regs(Bs[0]);

  for (int ki = 0; ki < nk; ++ki) {
    const int cur = ki & 1, nxt = cur ^ 1;
    const bool more = (ki + 1 < nk);
    if (more) {                       // issue next-tile global loads early
      loadA_regs((ki + 1) << 5);
      loadW_regs((ki + 1) << 5);
    }
    async_wait0();                    // this wave's async copies of tile ki done
    __syncthreads();                  // tile ki visible; buffer nxt free
    if (more) {
      issueA_async((ki + 1) << 5, As[nxt]);   // DMA overlaps compute below
      storeA_regs(As[nxt]);
      storeW_regs(Bs[nxt]);
    }
    v16bf af[2];
#pragma unroll
    for (int mi = 0; mi < 2; ++mi)
      af[mi] = load_a_frag(As[cur], wr * 32 + mi * 16, AP, 0, lane);
#pragma unroll
    for (int ni = 0; ni < 4; ++ni) {
      const v16bf bf = load_b_frag(Bs[cur], wc * 64 + ni * 16, BP, 0, lane);
#pragma unroll
      for (int mi = 0; mi < 2; ++mi) acc[mi][ni] = wmma_bf16(af[mi], bf, acc[mi][ni]);
    }
  }

  // ---- epilogue: (acc + bias) * scale, convert, store ----
  const int rsel = (lane >> 4) << 3;
  int bidx = 0, srow0 = 0;
  if constexpr (OUT_MODE == 1) { bidx = m0 / S; srow0 = m0 - bidx * S; }
#pragma unroll
  for (int ni = 0; ni < 4; ++ni) {
    const int n = n0 + wc * 64 + ni * 16 + (lane & 15);
    const float bv = bias[col_off + n];
    int h = 0, hd = 0;
    if constexpr (OUT_MODE == 1) { h = n / HDn; hd = n - h * HDn; }
#pragma unroll
    for (int mi = 0; mi < 2; ++mi) {
      const int moff = wr * 32 + mi * 16 + rsel;   // row offset within block tile
#pragma unroll
      for (int v = 0; v < 8; ++v) {
        const float val = (acc[mi][ni][v] + bv) * out_scale;
        if constexpr (OUT_MODE == 2) {
          ((float*)Cptr)[(size_t)(m0 + moff + v) * N + n] = val;
        } else if constexpr (OUT_MODE == 0) {
          ((unsigned short*)Cptr)[(size_t)(m0 + moff + v) * N + n] = f2bf(val);
        } else {
          const int s = srow0 + moff + v;
          ((unsigned short*)Cptr)[(((size_t)bidx * Hn + h) * S + s) * HDn + hd] =
              f2bf(val);
        }
      }
    }
  }
}

// ---------------------------------------------------------------------------
// Flash attention: per (b,h), 64-query tile per WG, stream 128-key chunks.
// Q,K,V bf16, head-reordered [B,H,S,64]; O bf16 [B,SQ,D]. Q pre-scaled by 1/sqrt(HD).
// Q/K tiles staged via async global->LDS DMA; V transposed through VGPRs.
// ---------------------------------------------------------------------------
__global__ __launch_bounds__(256) void attn_flash(
    const unsigned short* __restrict__ Qb, const unsigned short* __restrict__ Kb,
    const unsigned short* __restrict__ Vb, unsigned short* __restrict__ Ob,
    int SQ, int SK, int Hn, int Dm) {
  constexpr int HDc = 64, BQ = 64, BKC = 128;
  constexpr int QP = 72, KP = 72, VP = 136, PP = 132;
  __shared__ __align__(16) unsigned short Qs[BQ * QP];
  __shared__ __align__(16) unsigned short Ks[BKC * KP];
  __shared__ __align__(16) unsigned short Vst[HDc * VP];   // transposed [hd][key]
  __shared__ __align__(16) unsigned short Ps[BQ * PP];     // P in A layout [m][key]
  __shared__ float smax[2][BQ];
  __shared__ float ssum[2][BQ];

  const int tid = threadIdx.x, lane = tid & 31, wave = tid >> 5;
  const int mt  = wave >> 1;   // 0..3: 16-query band
  const int wcl = wave & 1;    // 0..1: key/col half
  const int q0  = blockIdx.x * BQ;
  const int bh  = blockIdx.y;
  const int b   = bh / Hn, h = bh % Hn;

  const unsigned short* Qg = Qb + ((size_t)bh * SQ + q0) * HDc;
  const unsigned short* Kg = Kb + (size_t)bh * SK * HDc;
  const unsigned short* Vg = Vb + (size_t)bh * SK * HDc;

  // stage Q tile (64x64) once, via async DMA (waited in first chunk)
#pragma unroll
  for (int i = 0; i < 2; ++i) {
    const int fl = tid + i * 256;          // 512 x 16B chunks
    const int row = fl >> 3, c = fl & 7;
    lds_copy16(Qg + row * HDc + c * 8, Qs + row * QP + c * 8);
  }

  const int rbase = mt * 16 + ((lane >> 4) << 3);  // first of this lane's 8 rows
  float mstate[8], lstate[8];
#pragma unroll
  for (int v = 0; v < 8; ++v) { mstate[v] = -1e30f; lstate[v] = 0.0f; }
  v8f oacc[2];
  oacc[0] = (v8f)0.0f; oacc[1] = (v8f)0.0f;

  for (int kc = 0; kc < SK; kc += BKC) {
    __syncthreads();   // previous chunk's readers of Ks/Vst/Ps are done
    // ---- stage K chunk (128x64) via async DMA ----
#pragma unroll
    for (int i = 0; i < 4; ++i) {
      const int fl = tid + i * 256;        // 1024 x 16B chunks
      const int row = fl >> 3, c = fl & 7;
      lds_copy16(Kg + (size_t)(kc + row) * HDc + c * 8, Ks + row * KP + c * 8);
    }
    // ---- stage V chunk (128x64) transposed -> Vst[hd][key] ----
#pragma unroll
    for (int i = 0; i < 16; ++i) {
      const int fl = tid + i * 256;        // 4096 bf16-pairs
      const int key = fl >> 5, hp = fl & 31;
      const uint32_t u =
          *reinterpret_cast<const uint32_t*>(Vg + (size_t)(kc + key) * HDc + hp * 2);
      Vst[(hp * 2 + 0) * VP + key] = (unsigned short)(u & 0xFFFFu);
      Vst[(hp * 2 + 1) * VP + key] = (unsigned short)(u >> 16);
    }
    if (kc + BKC < SK)   // warm L2 for next K chunk -> global_prefetch_b8
      __builtin_prefetch(Kg + (size_t)(kc + BKC + (tid >> 1)) * HDc + (tid & 1) * 32,
                         0, 1);
    async_wait0();
    __syncthreads();

    // ---- S = Q_scaled K^T: wave covers rows mt*16.., cols wcl*64.. ----
    v8f sacc[4];
#pragma unroll
    for (int ni = 0; ni < 4; ++ni) sacc[ni] = (v8f)0.0f;
#pragma unroll
    for (int ks = 0; ks < 2; ++ks) {                 // HD=64 -> 2 k-steps
      const v16bf aq = load_a_frag(Qs, mt * 16, QP, ks * 32, lane);
#pragma unroll
      for (int ni = 0; ni < 4; ++ni) {
        const v16bf bk = load_b_frag(Ks, wcl * 64 + ni * 16, KP, ks * 32, lane);
        sacc[ni] = wmma_bf16(aq, bk, sacc[ni]);
      }
    }
    float rmax[8];
#pragma unroll
    for (int v = 0; v < 8; ++v) rmax[v] = -1e30f;
#pragma unroll
    for (int ni = 0; ni < 4; ++ni)
#pragma unroll
      for (int v = 0; v < 8; ++v) rmax[v] = fmaxf(rmax[v], sacc[ni][v]);
#pragma unroll
    for (int off = 1; off < 16; off <<= 1)
#pragma unroll
      for (int v = 0; v < 8; ++v) rmax[v] = fmaxf(rmax[v], __shfl_xor(rmax[v], off, 32));
    if ((lane & 15) == 0)
#pragma unroll
      for (int v = 0; v < 8; ++v) smax[wcl][rbase + v] = rmax[v];
    __syncthreads();

    float mnew[8], alpha[8], rsum[8];
#pragma unroll
    for (int v = 0; v < 8; ++v) {
      const float cm = fmaxf(smax[0][rbase + v], smax[1][rbase + v]);
      mnew[v]  = fmaxf(mstate[v], cm);
      alpha[v] = __expf(mstate[v] - mnew[v]);
      mstate[v] = mnew[v];
      rsum[v] = 0.0f;
    }
    // ---- P = exp(S - mnew), stash bf16 into Ps (A layout) ----
#pragma unroll
    for (int ni = 0; ni < 4; ++ni) {
      const int kcol = wcl * 64 + ni * 16 + (lane & 15);
#pragma unroll
      for (int v = 0; v < 8; ++v) {
        const float p = __expf(sacc[ni][v] - mnew[v]);
        rsum[v] += p;
        Ps[(rbase + v) * PP + kcol] = f2bf(p);
      }
    }
#pragma unroll
    for (int off = 1; off < 16; off <<= 1)
#pragma unroll
      for (int v = 0; v < 8; ++v) rsum[v] += __shfl_xor(rsum[v], off, 32);
    if ((lane & 15) == 0)
#pragma unroll
      for (int v = 0; v < 8; ++v) ssum[wcl][rbase + v] = rsum[v];
    __syncthreads();

#pragma unroll
    for (int v = 0; v < 8; ++v)
      lstate[v] = alpha[v] * lstate[v] + ssum[0][rbase + v] + ssum[1][rbase + v];
#pragma unroll
    for (int t = 0; t < 2; ++t)
#pragma unroll
      for (int v = 0; v < 8; ++v) oacc[t][v] *= alpha[v];

    // ---- O += P * V  (k over 128 keys, wave cols wcl*32 + t*16) ----
#pragma unroll
    for (int ks = 0; ks < 4; ++ks) {
      const v16bf ap = load_a_frag(Ps, mt * 16, PP, ks * 32, lane);
#pragma unroll
      for (int t = 0; t < 2; ++t) {
        const v16bf bv = load_b_frag(Vst, wcl * 32 + t * 16, VP, ks * 32, lane);
        oacc[t] = wmma_bf16(ap, bv, oacc[t]);
      }
    }
  }

  // ---- normalize and write O (bf16, row-major [B,SQ,D]) ----
#pragma unroll
  for (int t = 0; t < 2; ++t) {
    const int col = h * HDc + wcl * 32 + t * 16 + (lane & 15);
#pragma unroll
    for (int v = 0; v < 8; ++v) {
      const int rq = q0 + rbase + v;
      const float val = oacc[t][v] / lstate[v];
      Ob[((size_t)b * SQ + rq) * Dm + col] = f2bf(val);
    }
  }
}

// ---------------------------------------------------------------------------
extern "C" void kernel_launch(void* const* d_in, const int* in_sizes, int n_in,
                              void* d_out, int out_size, void* d_ws, size_t ws_size,
                              hipStream_t stream) {
  (void)in_sizes; (void)n_in; (void)out_size; (void)ws_size;
  constexpr int B = 4, SQ = 2048, SK = 2048, D = 1024, H = 16, HD = 64;
  constexpr int M = B * SQ;   // 8192

  const float* query = (const float*)d_in[0];
  const float* keyv  = (const float*)d_in[1];
  const float* Wq    = (const float*)d_in[2];
  const float* bq    = (const float*)d_in[3];
  const float* Wkv   = (const float*)d_in[4];
  const float* bkv   = (const float*)d_in[5];
  const float* Wo    = (const float*)d_in[6];
  const float* bo    = (const float*)d_in[7];

  const size_t nElem = (size_t)B * H * SQ * HD;   // == 8M, same for K/V
  unsigned short* Qb = (unsigned short*)d_ws;
  unsigned short* Kb = Qb + nElem;
  unsigned short* Vb = Kb + nElem;
  unsigned short* Ob = Vb + nElem;

  const dim3 blk(256);
  const dim3 gP(D / 128, M / 128);   // 8 x 64

  // Q = (query@Wq + bq) * 1/sqrt(HD)  -> [B,H,SQ,HD] bf16  (scale folded in)
  gemm_proj<false, 1><<<gP, blk, 0, stream>>>(query, D, Wq, D, 0, bq, 0.125f, Qb,
                                              M, D, D, SQ, H, HD);
  // K = kv@Wkv[:, :D] + bkv[:D]  -> [B,H,SK,HD] bf16
  gemm_proj<false, 1><<<gP, blk, 0, stream>>>(keyv, D, Wkv, 2 * D, 0, bkv, 1.0f, Kb,
                                              M, D, D, SK, H, HD);
  // V = kv@Wkv[:, D:] + bkv[D:]  -> [B,H,SK,HD] bf16
  gemm_proj<false, 1><<<gP, blk, 0, stream>>>(keyv, D, Wkv, 2 * D, D, bkv, 1.0f, Vb,
                                              M, D, D, SK, H, HD);
  // attention -> Ob [B,SQ,D] bf16
  attn_flash<<<dim3(SQ / 64, B * H), blk, 0, stream>>>(Qb, Kb, Vb, Ob, SQ, SK, H, D);
  // out = Ob@Wo + bo -> f32 d_out
  gemm_proj<true, 2><<<gP, blk, 0, stream>>>(Ob, D, Wo, D, 0, bo, 1.0f, (float*)d_out,
                                             M, D, D, 1, 1, 1);
}
